// MultiHeadAttention_6193342841656
// MI455X (gfx1250) — matrix operand order repaired and checked by reading, not compiled, not run on hardware
//
#include <hip/hip_runtime.h>
#include <math.h>

// ---------------------------------------------------------------------------
// Transformer-XL relative attention block for MI455X (gfx1250, wave32, WMMA).
// B=8 S=512 MEM=512 T=1024 D=1024 H=16 DH=64.
// GEMMs: 8-wave workgroups, 64x128 tiles, TDM (tensor_load_to_lds) double-
// buffered LDS staging, v_wmma_f32_16x16x32_bf16 compute.
// Attention: fused per (b,h,16-row) tile, 4 waves, 128 KB LDS.
// ---------------------------------------------------------------------------

typedef __attribute__((ext_vector_type(16))) __bf16 v16bf;
typedef __attribute__((ext_vector_type(8)))  float  v8f;
typedef __attribute__((ext_vector_type(4)))  unsigned int u32x4;
typedef __attribute__((ext_vector_type(8)))  unsigned int u32x8;

#define BSZ   8
#define SLEN  512
#define MLEN  512
#define TLEN  1024
#define DMODEL 1024
#define NHEAD 16
#define DHEAD 64
#define HD    1024   // NHEAD*DHEAD

// ---- bf16 helpers (device only) -------------------------------------------
__device__ __forceinline__ __bf16 bits_to_bf(unsigned short u) {
  return __builtin_bit_cast(__bf16, u);
}
__device__ __forceinline__ unsigned short f32_to_bfbits(float f) {
  return __builtin_bit_cast(unsigned short, (__bf16)f);
}

// A-matrix 16x32 bf16 lane layout (ISA 7.12.2).
__device__ __forceinline__ int kmapA(int e, int khalf) {
  return ((e < 8) ? 0 : 16) + 8 * khalf + (e & 7);
}
// B-matrix 32x16 bf16: same K order as A (lanes 0-15 K=0..7,16..23; lanes 16-31 K=8..15,24..31).
__device__ __forceinline__ int kmapB(int e, int khalf) {
  return ((e < 8) ? 0 : 16) + 8 * khalf + (e & 7);
}

__device__ __forceinline__ v8f wmma_bf16(v16bf a, v16bf b, v8f c) {
  return __builtin_amdgcn_wmma_f32_16x16x32_bf16(
      false, a, false, b, (short)0, c, false, false);
}

// ---------------------------------------------------------------------------
// Tensor Data Mover: 2D tile load Global->LDS (ISA ch.8, D# groups 0/1).
// bf16 elements (data_size=1 -> 2 bytes).  Issued once per wave; tracked by
// TENSORcnt (s_wait_tensorcnt).
// ---------------------------------------------------------------------------
__device__ __forceinline__ void tdm_load_2d(unsigned lds_byte_off, const void* gptr,
                                            unsigned tile_cols, unsigned tile_rows,
                                            unsigned tensor_cols, unsigned tensor_rows,
                                            unsigned row_stride_elems) {
  unsigned long long ga = (unsigned long long)gptr;
  u32x4 g0;
  g0[0] = 1u;                                               // count=1, user D#
  g0[1] = lds_byte_off;                                     // lds_addr
  g0[2] = (unsigned)ga;                                     // global_addr[31:0]
  g0[3] = (unsigned)((ga >> 32) & 0x01ffffffu) | (2u << 30);// [56:32], type=2
  u32x8 g1;
  g1[0] = (1u << 16);                                       // data_size=1 (2B)
  g1[1] = (tensor_cols & 0xffffu) << 16;                    // tensor_dim0 lo
  g1[2] = (tensor_cols >> 16) | ((tensor_rows & 0xffffu) << 16);
  g1[3] = (tensor_rows >> 16) | (tile_cols << 16);          // tile_dim0
  g1[4] = tile_rows & 0xffffu;                              // tile_dim1
  g1[5] = row_stride_elems;                                 // dim0_stride lo32
  g1[6] = 0u;
  g1[7] = 0u;
  asm volatile("tensor_load_to_lds %0, %1" :: "s"(g0), "s"(g1) : "memory");
}

// ---------------------------------------------------------------------------
// fp32 -> bf16 conversion
// ---------------------------------------------------------------------------
__global__ void __launch_bounds__(256)
cvt_f32_bf16_kernel(const float* __restrict__ src, unsigned short* __restrict__ dst, int n) {
  int i = blockIdx.x * blockDim.x + threadIdx.x;
  if (i < n) dst[i] = f32_to_bfbits(src[i]);
}

// h = concat(mem, x) along seq, converted to bf16.  total = B*T*D = 2^23.
__global__ void __launch_bounds__(256)
concat_h_kernel(const float* __restrict__ mem, const float* __restrict__ x,
                unsigned short* __restrict__ h) {
  size_t i = (size_t)blockIdx.x * blockDim.x + threadIdx.x;
  size_t b  = i >> 20;               // T*D = 2^20
  size_t r  = i & ((1u << 20) - 1);
  size_t tt = r >> 10;               // D = 2^10
  size_t dd = r & 1023;
  float v = (tt < MLEN) ? mem[(b * MLEN + tt) * DMODEL + dd]
                        : x[(b * SLEN + (tt - MLEN)) * DMODEL + dd];
  h[i] = f32_to_bfbits(v);
}

// ---------------------------------------------------------------------------
// WMMA GEMM with TDM-staged LDS panels:
//   C[M,N] = A[M,K] * W[N,K]^T  (+ optional fp32 residual)
// Workgroup = 8 waves, 64(M) x 128(N) tile, BK=32, double-buffered LDS.
// Wave w: mw = w&3 (16 rows), nw = w>>2 (64 cols -> 4 WMMA accs).
// ---------------------------------------------------------------------------
template<bool OUT_BF16, bool RESIDUAL>
__global__ void __launch_bounds__(256)
wmma_gemm_kernel(const unsigned short* __restrict__ A,
                 const unsigned short* __restrict__ W,
                 float* __restrict__ Cf, unsigned short* __restrict__ Cb,
                 const float* __restrict__ res, int Mrows, int K, int N) {
  __shared__ unsigned short Ab[2][64 * 32];
  __shared__ unsigned short Bb[2][128 * 32];

  const int wid   = threadIdx.x >> 5;
  const int lane  = threadIdx.x & 31;
  const int khalf = lane >> 4;
  const int lrow  = lane & 15;
  const int mw    = wid & 3;
  const int nw    = wid >> 2;
  const size_t m0 = (size_t)blockIdx.y * 64;
  const size_t n0 = (size_t)blockIdx.x * 128;

  v8f acc[4] = {};

  if (wid == 0) {   // TDM ignores EXEC; one issue per wave
    tdm_load_2d((unsigned)(size_t)&Ab[0][0], A + m0 * K, 32, 64, K, Mrows, K);
    tdm_load_2d((unsigned)(size_t)&Bb[0][0], W + n0 * K, 32, 128, K, N, K);
    __builtin_amdgcn_s_wait_tensorcnt(0);
  }
  __syncthreads();

  for (int k0 = 0; k0 < K; k0 += 32) {
    const int cur  = (k0 >> 5) & 1;
    const int nxt  = cur ^ 1;
    const bool more = (k0 + 32) < K;
    if (more && wid == 0) {
      tdm_load_2d((unsigned)(size_t)&Ab[nxt][0], A + m0 * K + (k0 + 32), 32, 64, K, Mrows, K);
      tdm_load_2d((unsigned)(size_t)&Bb[nxt][0], W + n0 * K + (k0 + 32), 32, 128, K, N, K);
    }

    v16bf a;
#pragma unroll
    for (int e = 0; e < 16; ++e)
      a[e] = bits_to_bf(Ab[cur][(mw * 16 + lrow) * 32 + kmapA(e, khalf)]);
#pragma unroll
    for (int t4 = 0; t4 < 4; ++t4) {
      v16bf b;
#pragma unroll
      for (int e = 0; e < 16; ++e)
        b[e] = bits_to_bf(Bb[cur][(nw * 64 + t4 * 16 + lrow) * 32 + kmapB(e, khalf)]);
      acc[t4] = wmma_bf16(a, b, acc[t4]);
    }

    if (more && wid == 0) __builtin_amdgcn_s_wait_tensorcnt(0);
    __syncthreads();
  }

#pragma unroll
  for (int t4 = 0; t4 < 4; ++t4)
#pragma unroll
    for (int g = 0; g < 8; ++g) {
      size_t mm = m0 + mw * 16 + g + 8 * khalf;
      size_t nn = n0 + nw * 64 + t4 * 16 + lrow;
      float v = acc[t4][g];
      if (RESIDUAL) v += res[mm * N + nn];
      if (OUT_BF16) Cb[mm * N + nn] = f32_to_bfbits(v);
      else          Cf[mm * N + nn] = v;
    }
}

// ---------------------------------------------------------------------------
// Fused relative attention.  4 waves per (b, h, 16-row query tile).
// Dynamic LDS: bd[16][1024] fp32 + scores[16][1024] fp32 = 128 KB.
// ---------------------------------------------------------------------------
__global__ void __launch_bounds__(128)
attn_kernel(const unsigned short* __restrict__ qb,   // [B*S, HD] bf16
            const unsigned short* __restrict__ kb,   // [B*T, HD] bf16
            const unsigned short* __restrict__ vb,   // [B*T, HD] bf16
            const unsigned short* __restrict__ rb,   // [T,   HD] bf16
            const float* __restrict__ u,             // [HD]
            const float* __restrict__ t,             // [HD]
            unsigned short* __restrict__ attb) {     // [B*S, HD] bf16
  extern __shared__ float smem[];
  float* bdS = smem;               // [16][TLEN]
  float* scS = smem + 16 * TLEN;   // [16][TLEN]

  const int wid   = threadIdx.x >> 5;
  const int lane  = threadIdx.x & 31;
  const int khalf = lane >> 4;
  const int lrow  = lane & 15;

  const int bid  = blockIdx.x;
  const int tile = bid & 31;       // S/16 = 32 query tiles
  const int bh   = bid >> 5;
  const int b    = bh >> 4;        // NHEAD = 16
  const int hh   = bh & 15;
  const int i0   = tile * 16;

  // ---- build (q+u) and (q+t) A-fragments; K = DHEAD = 64 -> 2 frags each --
  v16bf qu[2], qt[2];
  {
    const unsigned short* qrow =
        qb + ((size_t)(b * SLEN + i0 + lrow)) * HD + hh * DHEAD;
    const float* uh = u + hh * DHEAD;
    const float* th = t + hh * DHEAD;
#pragma unroll
    for (int f = 0; f < 2; ++f) {
#pragma unroll
      for (int e = 0; e < 16; ++e) {
        int k = f * 32 + kmapA(e, khalf);
        float qv = (float)bits_to_bf(qrow[k]);
        qu[f][e] = (__bf16)(qv + uh[k]);
        qt[f][e] = (__bf16)(qv + th[k]);
      }
    }
  }

  // ---- Phase 1: bd_full[16][T] = (q+t) . r^T -> LDS (waves split jt) ------
  const unsigned short* rbase = rb + hh * DHEAD;
  for (int jt = wid * 16; jt < wid * 16 + 16; ++jt) {
    v8f acc = {};
#pragma unroll
    for (int f = 0; f < 2; ++f) {
      v16bf bf;
#pragma unroll
      for (int e = 0; e < 16; ++e)
        bf[e] = bits_to_bf(rbase[(size_t)(jt * 16 + lrow) * HD + f * 32 + kmapB(e, khalf)]);
      acc = wmma_bf16(qt[f], bf, acc);
    }
#pragma unroll
    for (int g = 0; g < 8; ++g)
      bdS[(g + 8 * khalf) * TLEN + jt * 16 + lrow] = acc[g];
  }
  __syncthreads();

  // ---- Phase 2: scores = mask((ac + shift(bd)) / sqrt(dh)) -> LDS ---------
  const unsigned short* kbase = kb + (size_t)b * TLEN * HD + hh * DHEAD;
  for (int jt = wid * 16; jt < wid * 16 + 16; ++jt) {
    v8f acc = {};
#pragma unroll
    for (int f = 0; f < 2; ++f) {
      v16bf bf;
#pragma unroll
      for (int e = 0; e < 16; ++e)
        bf[e] = bits_to_bf(kbase[(size_t)(jt * 16 + lrow) * HD + f * 32 + kmapB(e, khalf)]);
      acc = wmma_bf16(qu[f], bf, acc);
    }
#pragma unroll
    for (int g = 0; g < 8; ++g) {
      int m  = g + 8 * khalf;
      int ig = i0 + m;
      int j  = jt * 16 + lrow;
      int jr = (j - ig + SLEN - 1) & (TLEN - 1);     // circulant shift, T=2^10
      float s = (acc[g] + bdS[m * TLEN + jr]) * 0.125f;  // 1/sqrt(64)
      scS[m * TLEN + j] = (j <= ig + MLEN) ? s : -INFINITY;
    }
  }
  __syncthreads();

  // ---- Phase 3: row softmax; wave handles 4 rows, 8 lanes per row ---------
  {
    const int r  = wid * 4 + (lane >> 3);
    const int c0 = (lane & 7) * 128;
    float mx = -INFINITY;
    for (int j = c0; j < c0 + 128; ++j)
      mx = fmaxf(mx, scS[r * TLEN + j]);
    mx = fmaxf(mx, __shfl_xor(mx, 1));
    mx = fmaxf(mx, __shfl_xor(mx, 2));
    mx = fmaxf(mx, __shfl_xor(mx, 4));
    float sum = 0.f;
    for (int j = c0; j < c0 + 128; ++j)
      sum += __expf(scS[r * TLEN + j] - mx);
    sum += __shfl_xor(sum, 1);
    sum += __shfl_xor(sum, 2);
    sum += __shfl_xor(sum, 4);
    float inv = 1.0f / sum;
    for (int j = c0; j < c0 + 128; ++j)
      scS[r * TLEN + j] = __expf(scS[r * TLEN + j] - mx) * inv;
  }
  __syncthreads();

  // ---- Phase 4: att[16][64] = P[16][T] . V[T][64]; wave owns n-tile wid ---
  const unsigned short* vbase = vb + (size_t)b * TLEN * HD + hh * DHEAD;
  v8f acc = {};
  for (int kk = 0; kk < TLEN; kk += 32) {
    __builtin_prefetch(vbase + (size_t)(kk + 64) * HD + wid * 16 + lrow, 0, 0);
    v16bf af;
#pragma unroll
    for (int e = 0; e < 16; ++e)
      af[e] = (__bf16)scS[lrow * TLEN + kk + kmapA(e, khalf)];
    v16bf bf;
#pragma unroll
    for (int e = 0; e < 16; ++e)
      bf[e] = bits_to_bf(vbase[(size_t)(kk + kmapB(e, khalf)) * HD + wid * 16 + lrow]);
    acc = wmma_bf16(af, bf, acc);
  }
  unsigned short* arow = attb + (size_t)(b * SLEN + i0) * HD + hh * DHEAD;
#pragma unroll
  for (int g = 0; g < 8; ++g)
    arow[(size_t)(g + 8 * khalf) * HD + wid * 16 + lrow] = f32_to_bfbits(acc[g]);
}

// ---------------------------------------------------------------------------
// LayerNorm over last dim (D=1024), one block per row.
// ---------------------------------------------------------------------------
__global__ void __launch_bounds__(256)
layernorm_kernel(const float* __restrict__ y, const float* __restrict__ gamma,
                 const float* __restrict__ beta, float* __restrict__ out) {
  __shared__ float red[256];
  const int row = blockIdx.x;
  const int tid = threadIdx.x;
  const float* yr = y + (size_t)row * DMODEL;

  float s = 0.f;
  for (int j = tid; j < DMODEL; j += 256) s += yr[j];
  red[tid] = s; __syncthreads();
  for (int off = 128; off > 0; off >>= 1) {
    if (tid < off) red[tid] += red[tid + off];
    __syncthreads();
  }
  float mu = red[0] * (1.0f / DMODEL);
  __syncthreads();

  float v = 0.f;
  for (int j = tid; j < DMODEL; j += 256) {
    float d = yr[j] - mu;
    v += d * d;
  }
  red[tid] = v; __syncthreads();
  for (int off = 128; off > 0; off >>= 1) {
    if (tid < off) red[tid] += red[tid + off];
    __syncthreads();
  }
  float inv = rsqrtf(red[0] * (1.0f / DMODEL) + 1e-5f);

  float* orow = out + (size_t)row * DMODEL;
  for (int j = tid; j < DMODEL; j += 256)
    orow[j] = (yr[j] - mu) * inv * gamma[j] + beta[j];
}

// ---------------------------------------------------------------------------
// Launch
// ---------------------------------------------------------------------------
extern "C" void kernel_launch(void* const* d_in, const int* in_sizes, int n_in,
                              void* d_out, int out_size, void* d_ws, size_t ws_size,
                              hipStream_t stream) {
  (void)in_sizes; (void)n_in; (void)out_size; (void)ws_size;
  const float* x     = (const float*)d_in[0];
  const float* mem   = (const float*)d_in[1];
  const float* R     = (const float*)d_in[2];
  const float* u     = (const float*)d_in[3];
  const float* t     = (const float*)d_in[4];
  const float* Wq    = (const float*)d_in[5];
  const float* Wk    = (const float*)d_in[6];
  const float* Wv    = (const float*)d_in[7];
  const float* Wr    = (const float*)d_in[8];
  const float* Wo    = (const float*)d_in[9];
  const float* gamma = (const float*)d_in[10];
  const float* beta  = (const float*)d_in[11];
  float* out = (float*)d_out;

  const size_t BS = (size_t)BSZ * SLEN;   // 4096
  const size_t BT = (size_t)BSZ * TLEN;   // 8192

  char* ws = (char*)d_ws;
  size_t off = 0;
  auto alloc = [&](size_t bytes) {
    char* p = ws + off;
    off += (bytes + 255) & ~(size_t)255;
    return p;
  };
  unsigned short* xb   = (unsigned short*)alloc(BS * DMODEL * 2);
  unsigned short* hb   = (unsigned short*)alloc(BT * DMODEL * 2);
  unsigned short* Rb   = (unsigned short*)alloc((size_t)TLEN * DMODEL * 2);
  unsigned short* Wqb  = (unsigned short*)alloc((size_t)HD * DMODEL * 2);
  unsigned short* Wkb  = (unsigned short*)alloc((size_t)HD * DMODEL * 2);
  unsigned short* Wvb  = (unsigned short*)alloc((size_t)HD * DMODEL * 2);
  unsigned short* Wrb  = (unsigned short*)alloc((size_t)HD * DMODEL * 2);
  unsigned short* Wob  = (unsigned short*)alloc((size_t)DMODEL * HD * 2);
  unsigned short* qbuf = (unsigned short*)alloc(BS * HD * 2);
  unsigned short* kbuf = (unsigned short*)alloc(BT * HD * 2);
  unsigned short* vbuf = (unsigned short*)alloc(BT * HD * 2);
  unsigned short* rbuf = (unsigned short*)alloc((size_t)TLEN * HD * 2);
  unsigned short* attb = (unsigned short*)alloc(BS * HD * 2);
  float*          ybuf = (float*)alloc(BS * DMODEL * 4);

  // ---- conversions --------------------------------------------------------
  const int nXD = (int)(BS * DMODEL);       // 4M
  const int nWD = HD * DMODEL;              // 1M
  cvt_f32_bf16_kernel<<<(nXD + 255) / 256, 256, 0, stream>>>(x,  xb,  nXD);
  cvt_f32_bf16_kernel<<<(nWD + 255) / 256, 256, 0, stream>>>(R,  Rb,  nWD);
  cvt_f32_bf16_kernel<<<(nWD + 255) / 256, 256, 0, stream>>>(Wq, Wqb, nWD);
  cvt_f32_bf16_kernel<<<(nWD + 255) / 256, 256, 0, stream>>>(Wk, Wkb, nWD);
  cvt_f32_bf16_kernel<<<(nWD + 255) / 256, 256, 0, stream>>>(Wv, Wvb, nWD);
  cvt_f32_bf16_kernel<<<(nWD + 255) / 256, 256, 0, stream>>>(Wr, Wrb, nWD);
  cvt_f32_bf16_kernel<<<(nWD + 255) / 256, 256, 0, stream>>>(Wo, Wob, nWD);
  const int nHD = (int)(BT * DMODEL);       // 8M
  concat_h_kernel<<<(nHD + 255) / 256, 256, 0, stream>>>(mem, x, hb);

  // ---- projections (TDM + WMMA GEMMs), bf16 out ---------------------------
  dim3 blk(256, 1, 1);
  dim3 gq(HD / 128, (unsigned)(BS / 64));   // 8 x 64
  dim3 gk(HD / 128, (unsigned)(BT / 64));   // 8 x 128
  dim3 gr(HD / 128, TLEN / 64);             // 8 x 16
  wmma_gemm_kernel<true, false><<<gq, blk, 0, stream>>>(xb, Wqb, nullptr, qbuf, nullptr, (int)BS, DMODEL, HD);
  wmma_gemm_kernel<true, false><<<gk, blk, 0, stream>>>(hb, Wkb, nullptr, kbuf, nullptr, (int)BT, DMODEL, HD);
  wmma_gemm_kernel<true, false><<<gk, blk, 0, stream>>>(hb, Wvb, nullptr, vbuf, nullptr, (int)BT, DMODEL, HD);
  wmma_gemm_kernel<true, false><<<gr, blk, 0, stream>>>(Rb, Wrb, nullptr, rbuf, nullptr, TLEN, DMODEL, HD);

  // ---- fused attention ----------------------------------------------------
  const int smem_bytes = 2 * 16 * TLEN * (int)sizeof(float);   // 128 KB
  hipFuncSetAttribute((const void*)attn_kernel,
                      hipFuncAttributeMaxDynamicSharedMemorySize, smem_bytes);
  attn_kernel<<<BSZ * NHEAD * (SLEN / 16), 128, smem_bytes, stream>>>(
      qbuf, kbuf, vbuf, rbuf, u, t, attb);

  // ---- output projection + residual, then LayerNorm -----------------------
  wmma_gemm_kernel<false, true><<<gq, blk, 0, stream>>>(attb, Wob, ybuf, nullptr, x, (int)BS, HD, DMODEL);
  layernorm_kernel<<<(unsigned)BS, 256, 0, stream>>>(ybuf, gamma, beta, out);
}
